// GCM_23390391894818
// MI455X (gfx1250) — matrix operand order repaired and checked by reading, not compiled, not run on hardware
//
#include <hip/hip_runtime.h>
#include <hip/hip_bf16.h>
#include <math.h>

// ---------------- problem constants ----------------
#define CCH  2304                 // channels (== tokens after pool)
#define HP   48
#define WP   48
#define NTOK (HP*WP)              // 2304
#define HF   96
#define WF   96
#define TTOK (HF*WF)              // 9216

typedef __bf16 bf16_t;
typedef bf16_t bf16x16 __attribute__((ext_vector_type(16)));
typedef float  f32x8  __attribute__((ext_vector_type(8)));

__device__ __forceinline__ float gelu_exact(float v) {
    return 0.5f * v * (1.0f + erff(v * 0.70710678118654752440f));
}

// ============================================================
// WMMA bf16 GEMM: D[m,o] = alpha*acc*cs[o] + cb[o] (+res[m,o])
//   A  : [M,K]  bf16 row-major (K contiguous)
//   Bt : [Nc,K] bf16 row-major (K contiguous)  (i.e. B transposed)
// Tile: 128x128 per block, 256 threads = 8 wave32, BK=64.
// Double-buffered LDS, filled with CDNA5 async global->LDS copies
// (ASYNCcnt) so HBM/L2 latency overlaps the WMMA stream.
// ============================================================
#define BK    64
#define LDST  72   // padded LDS row stride (bf16 elems) -> conflict-free frag reads

__global__ __launch_bounds__(256) void k_gemm(
    const bf16_t* __restrict__ A, const bf16_t* __restrict__ Bt,
    int M, int Nc, int K, float alpha,
    const float* __restrict__ cs, const float* __restrict__ cb,
    const float* __restrict__ res,
    float* __restrict__ outF, int oft,
    bf16_t* __restrict__ outH, int oht)
{
    __shared__ bf16_t sA[2][128 * LDST];
    __shared__ bf16_t sB[2][128 * LDST];

    const int tid  = threadIdx.x;
    const int lane = tid & 31;
    const int l15  = lane & 15;
    const int half = lane >> 4;
    const int wave = tid >> 5;
    const int wm   = wave >> 1;   // 0..3 : 32-row strip
    const int wn   = wave & 1;    // 0..1 : 64-col strip
    const int m0   = blockIdx.y * 128;
    const int n0   = blockIdx.x * 128;

    f32x8 acc[2][4] = {};

    // Issue async global->LDS copies for k-tile kb into buffer buf.
    // 256 threads x 4 chunks cover each 128x64 bf16 tile of A and Bt.
    auto stage_tile = [&](int kb, int buf) {
        const int k0 = kb * BK;
        #pragma unroll
        for (int i = 0; i < 4; ++i) {
            int idx  = tid + (i << 8);     // 0..1023
            int row  = idx >> 3;           // 0..127
            int col8 = (idx & 7) << 3;     // 0,8,..,56
            uint64_t ga = (uint64_t)(uintptr_t)(A  + (size_t)(m0 + row) * K + k0 + col8);
            uint64_t gb = (uint64_t)(uintptr_t)(Bt + (size_t)(n0 + row) * K + k0 + col8);
            uint32_t la = (uint32_t)(uintptr_t)(&sA[buf][row * LDST + col8]);
            uint32_t lb = (uint32_t)(uintptr_t)(&sB[buf][row * LDST + col8]);
            asm volatile("global_load_async_to_lds_b128 %0, %1, off"
                         :: "v"(la), "v"(ga) : "memory");
            asm volatile("global_load_async_to_lds_b128 %0, %1, off"
                         :: "v"(lb), "v"(gb) : "memory");
        }
    };

    // 16 WMMAs over one 128x64 * 64x128 LDS tile pair.
    auto compute_tile = [&](const bf16_t* bA, const bf16_t* bB) {
        #pragma unroll
        for (int ks = 0; ks < BK; ks += 32) {
            // A fragments (16x32 bf16): lane l15=row, v0-3 = K[h*8..], v4-7 = K[16+h*8..]
            bf16x16 af[2];
            #pragma unroll
            for (int i = 0; i < 2; ++i) {
                const bf16_t* p = bA + (wm * 32 + i * 16 + l15) * LDST + ks + half * 8;
                union { uint4 u[2]; bf16x16 v; } t;
                t.u[0] = *(const uint4*)(p);
                t.u[1] = *(const uint4*)(p + 16);
                af[i] = t.v;
            }
            // B fragments: lane l15=col, v0-3 = K[h*8..], v4-7 = K[16+h*8..] (same order as A)
            #pragma unroll
            for (int j = 0; j < 4; ++j) {
                const bf16_t* p = bB + (wn * 64 + j * 16 + l15) * LDST + ks + half * 8;
                union { uint4 u[2]; bf16x16 v; } t;
                t.u[0] = *(const uint4*)(p);
                t.u[1] = *(const uint4*)(p + 16);
                bf16x16 bfj = t.v;
                #pragma unroll
                for (int i = 0; i < 2; ++i) {
                    acc[i][j] = __builtin_amdgcn_wmma_f32_16x16x32_bf16(
                        false, af[i], false, bfj, (short)0, acc[i][j], false, false);
                }
            }
        }
    };

    #define WAIT_ASYNC() asm volatile("s_wait_asynccnt 0" ::: "memory")

    const int nk = K / BK;
    stage_tile(0, 0);
    WAIT_ASYNC();
    __syncthreads();

    for (int kb = 0; kb < nk; kb += 2) {
        if (kb + 1 < nk) stage_tile(kb + 1, 1);    // prefetch next tile while computing
        compute_tile(sA[0], sB[0]);
        WAIT_ASYNC();
        __syncthreads();
        if (kb + 1 < nk) {
            if (kb + 2 < nk) stage_tile(kb + 2, 0);
            compute_tile(sA[1], sB[1]);
            WAIT_ASYNC();
            __syncthreads();
        }
    }
    #undef WAIT_ASYNC

    // -------- epilogue: C/D layout: VGPR r, M = r + 8*half, N = l15 --------
    #pragma unroll
    for (int i = 0; i < 2; ++i) {
        #pragma unroll
        for (int j = 0; j < 4; ++j) {
            const int o   = n0 + wn * 64 + j * 16 + l15;
            const float scl = alpha * (cs ? cs[o] : 1.0f);
            const float sh  = cb ? cb[o] : 0.0f;
            #pragma unroll
            for (int r = 0; r < 8; ++r) {
                const int m = m0 + wm * 32 + i * 16 + r + half * 8;
                float v = acc[i][j][r] * scl + sh;
                if (res)  v += res[(size_t)m * Nc + o];
                if (outF) {
                    if (oft) outF[(size_t)o * M + m] = v;
                    else     outF[(size_t)m * Nc + o] = v;
                }
                if (outH) {
                    if (oht) outH[(size_t)o * M + m] = (bf16_t)v;
                    else     outH[(size_t)m * Nc + o] = (bf16_t)v;
                }
            }
        }
    }
}

// ---------------- elementwise / reduction kernels ----------------

// AvgPool2d(2,2): [C,96,96] -> token-major [N,C] (f32 + bf16)
__global__ void k_avgpool(const float* __restrict__ x,
                          float* __restrict__ xf, bf16_t* __restrict__ xh) {
    const int t = blockIdx.x;
    const int r = t / WP, q = t - r * WP;
    const int r2 = 2 * r, q2 = 2 * q;
    for (int c = threadIdx.x; c < CCH; c += blockDim.x) {
        const float* p = x + (size_t)c * (HF * WF);
        float v = 0.25f * (p[r2 * WF + q2]     + p[r2 * WF + q2 + 1] +
                           p[(r2+1) * WF + q2] + p[(r2+1) * WF + q2 + 1]);
        xf[(size_t)t * CCH + c] = v;
        xh[(size_t)t * CCH + c] = (bf16_t)v;
    }
}

// per-channel mean over tokens
__global__ void k_colmean(const float* __restrict__ xf, float* __restrict__ meanv) {
    int c = blockIdx.x * blockDim.x + threadIdx.x;
    if (c >= CCH) return;
    float s = 0.f;
    for (int t = 0; t < NTOK; ++t) s += xf[(size_t)t * CCH + c];
    meanv[c] = s * (1.0f / NTOK);
}

// xg = mean[c] * x
__global__ void k_xg(const float* __restrict__ xf, const float* __restrict__ meanv,
                     float* __restrict__ xgf, bf16_t* __restrict__ xgh) {
    size_t tot = (size_t)NTOK * CCH;
    for (size_t i = blockIdx.x * (size_t)blockDim.x + threadIdx.x; i < tot;
         i += (size_t)gridDim.x * blockDim.x) {
        int c = (int)(i % CCH);
        float v = meanv[c] * xf[i];
        xgf[i] = v; xgh[i] = (bf16_t)v;
    }
}

__global__ void k_f2b(const float* __restrict__ s, bf16_t* __restrict__ d, size_t n) {
    for (size_t i = blockIdx.x * (size_t)blockDim.x + threadIdx.x; i < n;
         i += (size_t)gridDim.x * blockDim.x)
        d[i] = (bf16_t)s[i];
}

// xv = sigmoid(xq)*xk + xg   (bf16 out)
__global__ void k_xv(const float* __restrict__ xq, const float* __restrict__ xk,
                     const float* __restrict__ xg, bf16_t* __restrict__ xv) {
    size_t tot = (size_t)NTOK * CCH;
    for (size_t i = blockIdx.x * (size_t)blockDim.x + threadIdx.x; i < tot;
         i += (size_t)gridDim.x * blockDim.x) {
        float s = 1.0f / (1.0f + expf(-xq[i]));
        xv[i] = (bf16_t)(s * xk[i] + xg[i]);
    }
}

// depthwise 5-tap conv (dir=0 vertical 5x1, dir=1 horizontal 1x5) + GELU, token-major
__global__ void k_dw5(const float* __restrict__ src, const float* __restrict__ w,
                      const float* __restrict__ b, int dir, bf16_t* __restrict__ out) {
    const int t = blockIdx.x;
    const int r = t / WP, q = t - r * WP;
    for (int c = threadIdx.x; c < CCH; c += blockDim.x) {
        float acc = b[c];
        #pragma unroll
        for (int i = 0; i < 5; ++i) {
            int rr = r, qq = q;
            if (dir == 0) rr = r + i - 2; else qq = q + i - 2;
            if (rr >= 0 && rr < HP && qq >= 0 && qq < WP)
                acc += w[c * 5 + i] * src[(size_t)(rr * WP + qq) * CCH + c];
        }
        out[(size_t)t * CCH + c] = (bf16_t)gelu_exact(acc);
    }
}

// depthwise 3x3 conv + GELU on the 96x96 grid, token-major
__global__ void k_dw3(const float* __restrict__ src, const float* __restrict__ w,
                      const float* __restrict__ b, bf16_t* __restrict__ out) {
    const int t = blockIdx.x;
    const int r = t / WF, q = t - r * WF;
    for (int c = threadIdx.x; c < CCH; c += blockDim.x) {
        float acc = b[c];
        #pragma unroll
        for (int dy = 0; dy < 3; ++dy) {
            int rr = r + dy - 1;
            if (rr < 0 || rr >= HF) continue;
            #pragma unroll
            for (int dx = 0; dx < 3; ++dx) {
                int qq = q + dx - 1;
                if (qq < 0 || qq >= WF) continue;
                acc += w[c * 9 + dy * 3 + dx] * src[(size_t)(rr * WF + qq) * CCH + c];
            }
        }
        out[(size_t)t * CCH + c] = (bf16_t)gelu_exact(acc);
    }
}

// fold pointwise bias + BN into per-column scale/shift
__global__ void k_bnprep(const float* __restrict__ g, const float* __restrict__ b,
                         const float* __restrict__ m, const float* __restrict__ v,
                         const float* __restrict__ pwb,
                         float* __restrict__ cs, float* __restrict__ cb) {
    int c = blockIdx.x * blockDim.x + threadIdx.x;
    if (c >= CCH) return;
    float s = g[c] * rsqrtf(v[c] + 1e-5f);
    cs[c] = s;
    cb[c] = (pwb[c] - m[c]) * s + b[c];
}

// softmax row stats of S (for A_hl)
__global__ void k_rowstats(const float* __restrict__ S,
                           float* __restrict__ rmax, float* __restrict__ rinv) {
    __shared__ float red[256];
    const int n = blockIdx.x, tid = threadIdx.x;
    const float* row = S + (size_t)n * NTOK;
    float mx = -3.402823466e38f;
    for (int j = tid; j < NTOK; j += 256) mx = fmaxf(mx, row[j]);
    red[tid] = mx; __syncthreads();
    for (int s = 128; s > 0; s >>= 1) { if (tid < s) red[tid] = fmaxf(red[tid], red[tid + s]); __syncthreads(); }
    mx = red[0]; __syncthreads();
    float sm = 0.f;
    for (int j = tid; j < NTOK; j += 256) sm += expf(row[j] - mx);
    red[tid] = sm; __syncthreads();
    for (int s = 128; s > 0; s >>= 1) { if (tid < s) red[tid] += red[tid + s]; __syncthreads(); }
    if (tid == 0) { rmax[n] = mx; rinv[n] = 1.0f / red[0]; }
}

// softmax column stats of S (for A_lh)
__global__ void k_colstats(const float* __restrict__ S,
                           float* __restrict__ cmax, float* __restrict__ cinv) {
    int c = blockIdx.x * blockDim.x + threadIdx.x;
    if (c >= NTOK) return;
    float mx = -3.402823466e38f;
    for (int r = 0; r < NTOK; ++r) mx = fmaxf(mx, S[(size_t)r * NTOK + c]);
    float sm = 0.f;
    for (int r = 0; r < NTOK; ++r) sm += expf(S[(size_t)r * NTOK + c] - mx);
    cmax[c] = mx; cinv[c] = 1.0f / sm;
}

// Msum[n,m] = A_hl[n,m] + A_lh[n,m]  (bf16)
__global__ void k_msum(const float* __restrict__ S,
                       const float* __restrict__ rmax, const float* __restrict__ rinv,
                       const float* __restrict__ cmax, const float* __restrict__ cinv,
                       bf16_t* __restrict__ Msum) {
    const int n = blockIdx.x;
    const float rm = rmax[n], ri = rinv[n], cm = cmax[n], ci = cinv[n];
    for (int m = threadIdx.x; m < NTOK; m += blockDim.x) {
        float a = expf(S[(size_t)n * NTOK + m] - rm) * ri;          // row-softmax of S
        float b = expf(S[(size_t)m * NTOK + n] - cm) * ci;          // col-softmax of S, transposed
        Msum[(size_t)n * NTOK + m] = (bf16_t)(a + b);
    }
}

// v[m] = mean_n Msum[n,m]
__global__ void k_colmean_h(const bf16_t* __restrict__ Msum, float* __restrict__ v) {
    int m = blockIdx.x * blockDim.x + threadIdx.x;
    if (m >= NTOK) return;
    float s = 0.f;
    for (int n = 0; n < NTOK; ++n) s += (float)Msum[(size_t)n * NTOK + m];
    v[m] = s * (1.0f / NTOK);
}

// dyn = v @ W^T + b  (block per output)
__global__ void k_matvec(const float* __restrict__ W, const float* __restrict__ b,
                         const float* __restrict__ v, float* __restrict__ dyn) {
    __shared__ float red[256];
    const int o = blockIdx.x, tid = threadIdx.x;
    const float* row = W + (size_t)o * CCH;
    float s = 0.f;
    for (int k = tid; k < CCH; k += 256) s += row[k] * v[k];
    red[tid] = s; __syncthreads();
    for (int st = 128; st > 0; st >>= 1) { if (tid < st) red[tid] += red[tid + st]; __syncthreads(); }
    if (tid == 0) dyn[o] = red[0] + b[o];
}

// prompt = dyn*P + xV ; LayerNorm ; + prompt_l + prompt_h  (block per token)
__global__ void k_promptln(const float* __restrict__ P, const float* __restrict__ xV,
                           const float* __restrict__ dyn,
                           const float* __restrict__ lg, const float* __restrict__ lb,
                           const float* __restrict__ pl, const float* __restrict__ ph,
                           float* __restrict__ pn) {
    __shared__ float sp[CCH];
    __shared__ float r1[256];
    __shared__ float r2[256];
    const int n = blockIdx.x, tid = threadIdx.x;
    float s1 = 0.f, s2 = 0.f;
    for (int c = tid; c < CCH; c += 256) {
        float p = dyn[c] * P[(size_t)n * CCH + c] + xV[(size_t)n * CCH + c];
        sp[c] = p; s1 += p; s2 += p * p;
    }
    r1[tid] = s1; r2[tid] = s2; __syncthreads();
    for (int st = 128; st > 0; st >>= 1) {
        if (tid < st) { r1[tid] += r1[tid + st]; r2[tid] += r2[tid + st]; }
        __syncthreads();
    }
    const float mu = r1[0] * (1.0f / CCH);
    const float var = r2[0] * (1.0f / CCH) - mu * mu;
    const float rstd = rsqrtf(var + 1e-5f);
    for (int c = tid; c < CCH; c += 256) {
        float v = (sp[c] - mu) * rstd * lg[c] + lb[c]
                + pl[(size_t)n * CCH + c] + ph[(size_t)n * CCH + c];
        pn[(size_t)n * CCH + c] = v;
    }
}

// bilinear x2 upsample, align_corners=True; token-major 48x48 -> 96x96
__global__ void k_up2(const float* __restrict__ pn, float* __restrict__ up) {
    const int t = blockIdx.x;
    const int yy = t / WF, xx = t - yy * WF;
    const float fy = yy * (47.0f / 95.0f);
    const float fx = xx * (47.0f / 95.0f);
    const int y0 = (int)fy, x0 = (int)fx;
    const int y1 = min(y0 + 1, HP - 1), x1 = min(x0 + 1, WP - 1);
    const float wy = fy - y0, wx = fx - x0;
    const float w00 = (1.f - wy) * (1.f - wx), w01 = (1.f - wy) * wx;
    const float w10 = wy * (1.f - wx),         w11 = wy * wx;
    const size_t t00 = (size_t)(y0 * WP + x0) * CCH, t01 = (size_t)(y0 * WP + x1) * CCH;
    const size_t t10 = (size_t)(y1 * WP + x0) * CCH, t11 = (size_t)(y1 * WP + x1) * CCH;
    for (int c = threadIdx.x; c < CCH; c += blockDim.x) {
        up[(size_t)t * CCH + c] = w00 * pn[t00 + c] + w01 * pn[t01 + c]
                                + w10 * pn[t10 + c] + w11 * pn[t11 + c];
    }
}

// ---------------- host-side launch ----------------
static inline void gemm_launch(hipStream_t s, const bf16_t* A, const bf16_t* Bt,
                               int M, int Nc, int K, float alpha,
                               const float* cs, const float* cb, const float* res,
                               float* outF, int oft, bf16_t* outH, int oht) {
    dim3 g(Nc / 128, M / 128);
    k_gemm<<<g, 256, 0, s>>>(A, Bt, M, Nc, K, alpha, cs, cb, res, outF, oft, outH, oht);
}

extern "C" void kernel_launch(void* const* d_in, const int* in_sizes, int n_in,
                              void* d_out, int out_size, void* d_ws, size_t ws_size,
                              hipStream_t stream) {
    (void)in_sizes; (void)n_in; (void)out_size; (void)ws_size;
    // ---- input map (setup_inputs dict order) ----
    const float* x        = (const float*)d_in[0];
    const float* lin_o_w  = (const float*)d_in[1];
    const float* lin_o_b  = (const float*)d_in[2];
    const float* lin_p_w  = (const float*)d_in[3];
    const float* lin_p_b  = (const float*)d_in[4];
    const float* q_w      = (const float*)d_in[5];
    const float* q_b      = (const float*)d_in[6];
    const float* k_w      = (const float*)d_in[7];
    const float* k_b      = (const float*)d_in[8];
    const float* lin_l_w  = (const float*)d_in[9];
    const float* lin_l_b  = (const float*)d_in[10];
    const float* ln_g     = (const float*)d_in[11];
    const float* ln_b     = (const float*)d_in[12];
    const float* dsc_dw_w = (const float*)d_in[13];
    const float* dsc_dw_b = (const float*)d_in[14];
    const float* dsc_pw_w = (const float*)d_in[15];
    const float* dsc_pw_b = (const float*)d_in[16];
    const float* dsc_bn_g = (const float*)d_in[17];
    const float* dsc_bn_b = (const float*)d_in[18];
    const float* dsc_bn_m = (const float*)d_in[19];
    const float* dsc_bn_v = (const float*)d_in[20];
    const float* sch_dw_w = (const float*)d_in[21];
    const float* sch_dw_b = (const float*)d_in[22];
    const float* sch_pw_w = (const float*)d_in[23];
    const float* sch_pw_b = (const float*)d_in[24];
    const float* sch_bn_g = (const float*)d_in[25];
    const float* sch_bn_b = (const float*)d_in[26];
    const float* sch_bn_m = (const float*)d_in[27];
    const float* sch_bn_v = (const float*)d_in[28];
    const float* scv_dw_w = (const float*)d_in[29];
    const float* scv_dw_b = (const float*)d_in[30];
    const float* scv_pw_w = (const float*)d_in[31];
    const float* scv_pw_b = (const float*)d_in[32];
    const float* scv_bn_g = (const float*)d_in[33];
    const float* scv_bn_b = (const float*)d_in[34];
    const float* scv_bn_m = (const float*)d_in[35];
    const float* scv_bn_v = (const float*)d_in[36];
    const float* convh_w  = (const float*)d_in[37];
    const float* convh_b  = (const float*)d_in[38];

    // ---- workspace layout (bump allocator, 256B aligned) ----
    char* W = (char*)d_ws;
    size_t off = 0;
    auto bump = [&](size_t sz) -> char* {
        char* p = W + off; off += (sz + 255) & ~(size_t)255; return p;
    };
    const size_t F  = (size_t)NTOK * CCH * sizeof(float);
    const size_t Hb = (size_t)NTOK * CCH * sizeof(bf16_t);

    float*  xtok_f = (float*) bump(F);
    bf16_t* xtok_h = (bf16_t*)bump(Hb);
    float*  meanv  = (float*) bump(CCH * 4);
    float*  xg_f   = (float*) bump(F);
    bf16_t* xg_h   = (bf16_t*)bump(Hb);
    float*  xq     = (float*) bump(F);
    float*  xk     = (float*) bump(F);
    bf16_t* xv_h   = (bf16_t*)bump(Hb);
    float*  pl_f   = (float*) bump(F);
    bf16_t* pl_h   = (bf16_t*)bump(Hb);
    bf16_t* dw_h   = (bf16_t*)bump(Hb);
    float*  xh_f   = (float*) bump(F);
    bf16_t* hsum_h = (bf16_t*)bump(Hb);
    float*  ph_f   = (float*) bump(F);
    bf16_t* ph_h   = (bf16_t*)bump(Hb);
    float*  xV_f   = (float*) bump(F);
    bf16_t* xVt_h  = (bf16_t*)bump(Hb);
    float*  rmax   = (float*) bump(CCH * 4);
    float*  rinv   = (float*) bump(CCH * 4);
    float*  cmax   = (float*) bump(CCH * 4);
    float*  cinv   = (float*) bump(CCH * 4);
    float*  vvec   = (float*) bump(CCH * 4);
    float*  dynv   = (float*) bump(CCH * 4);
    float*  csv    = (float*) bump(CCH * 4);
    float*  cbv    = (float*) bump(CCH * 4);
    bf16_t* wbuf   = (bf16_t*)bump(Hb);

    // lifetime-safe aliases
    float*  S    = xq;                 // xq dead after k_xv
    float*  P    = xk;                 // xk dead after k_xv
    float*  pn   = xg_f;               // xg_f dead after k_xv
    bf16_t* Msum = xg_h;               // xg_h dead after q/k GEMMs
    float*  up   = xq;                 // 4F span == xq..pl_h (exactly 3F+2H), all dead at up2
    bf16_t* g_h  = (bf16_t*)(void*)xh_f; // 2F span within xh..ph_f, all dead at dw3

    const size_t wcnt = (size_t)CCH * CCH;
    const float attn_scale = 1.0f / 48.0f;   // 2304^-0.5

    // ---- stage 0: pool + stats ----
    k_avgpool<<<NTOK, 256, 0, stream>>>(x, xtok_f, xtok_h);
    k_colmean<<<(CCH + 255) / 256, 256, 0, stream>>>(xtok_f, meanv);
    k_xg<<<2048, 256, 0, stream>>>(xtok_f, meanv, xg_f, xg_h);

    // ---- LPG ----
    k_f2b<<<2048, 256, 0, stream>>>(q_w, wbuf, wcnt);
    gemm_launch(stream, xg_h, wbuf, NTOK, CCH, CCH, 1.f, nullptr, q_b, nullptr, xq, 0, nullptr, 0);
    k_f2b<<<2048, 256, 0, stream>>>(k_w, wbuf, wcnt);
    gemm_launch(stream, xg_h, wbuf, NTOK, CCH, CCH, 1.f, nullptr, k_b, nullptr, xk, 0, nullptr, 0);
    k_xv<<<2048, 256, 0, stream>>>(xq, xk, xg_f, xv_h);
    k_f2b<<<2048, 256, 0, stream>>>(lin_l_w, wbuf, wcnt);
    gemm_launch(stream, xv_h, wbuf, NTOK, CCH, CCH, 1.f, nullptr, lin_l_b, nullptr, pl_f, 0, pl_h, 0);

    // ---- xV = lin_o(x); also emit xV^T in bf16 for the P GEMM ----
    k_f2b<<<2048, 256, 0, stream>>>(lin_o_w, wbuf, wcnt);
    gemm_launch(stream, xtok_h, wbuf, NTOK, CCH, CCH, 1.f, nullptr, lin_o_b, nullptr, xV_f, 0, xVt_h, 1);

    // ---- HPG ----
    k_dw5<<<NTOK, 256, 0, stream>>>(xtok_f, sch_dw_w, sch_dw_b, 0, dw_h);
    k_bnprep<<<(CCH + 255) / 256, 256, 0, stream>>>(sch_bn_g, sch_bn_b, sch_bn_m, sch_bn_v, sch_pw_b, csv, cbv);
    k_f2b<<<2048, 256, 0, stream>>>(sch_pw_w, wbuf, wcnt);
    gemm_launch(stream, dw_h, wbuf, NTOK, CCH, CCH, 1.f, csv, cbv, nullptr, xh_f, 0, nullptr, 0);
    k_dw5<<<NTOK, 256, 0, stream>>>(xtok_f, scv_dw_w, scv_dw_b, 1, dw_h);
    k_bnprep<<<(CCH + 255) / 256, 256, 0, stream>>>(scv_bn_g, scv_bn_b, scv_bn_m, scv_bn_v, scv_pw_b, csv, cbv);
    k_f2b<<<2048, 256, 0, stream>>>(scv_pw_w, wbuf, wcnt);
    gemm_launch(stream, dw_h, wbuf, NTOK, CCH, CCH, 1.f, csv, cbv, xh_f, nullptr, 0, hsum_h, 0); // hsum = BN(xvv)+xh
    k_f2b<<<2048, 256, 0, stream>>>(convh_w, wbuf, wcnt);
    gemm_launch(stream, hsum_h, wbuf, NTOK, CCH, CCH, 1.f, nullptr, convh_b, nullptr, ph_f, 0, ph_h, 0);

    // ---- PGC: S = Qh Kl^T * scale ----
    gemm_launch(stream, ph_h, pl_h, NTOK, NTOK, CCH, attn_scale, nullptr, nullptr, nullptr, S, 0, nullptr, 0);
    k_rowstats<<<NTOK, 256, 0, stream>>>(S, rmax, rinv);
    k_colstats<<<(NTOK + 255) / 256, 256, 0, stream>>>(S, cmax, cinv);
    k_msum<<<NTOK, 256, 0, stream>>>(S, rmax, rinv, cmax, cinv, Msum);
    k_colmean_h<<<(NTOK + 255) / 256, 256, 0, stream>>>(Msum, vvec);
    k_matvec<<<CCH, 256, 0, stream>>>(lin_p_w, lin_p_b, vvec, dynv);
    gemm_launch(stream, Msum, xVt_h, NTOK, CCH, NTOK, 1.f, nullptr, nullptr, nullptr, P, 0, nullptr, 0);
    k_promptln<<<NTOK, 256, 0, stream>>>(P, xV_f, dynv, ln_g, ln_b, pl_f, ph_f, pn);
    k_up2<<<TTOK, 256, 0, stream>>>(pn, up);

    // ---- dsc block + final residual, stored transposed straight into d_out [C,96,96] ----
    k_dw3<<<TTOK, 256, 0, stream>>>(up, dsc_dw_w, dsc_dw_b, g_h);
    k_bnprep<<<(CCH + 255) / 256, 256, 0, stream>>>(dsc_bn_g, dsc_bn_b, dsc_bn_m, dsc_bn_v, dsc_pw_b, csv, cbv);
    k_f2b<<<2048, 256, 0, stream>>>(dsc_pw_w, wbuf, wcnt);
    gemm_launch(stream, g_h, wbuf, TTOK, CCH, CCH, 1.f, csv, cbv, up, (float*)d_out, 1, nullptr, 0);
}
